// MultiheadAttention_72971494359421
// MI455X (gfx1250) — matrix operand order repaired and checked by reading, not compiled, not run on hardware
//
#include <hip/hip_runtime.h>
#include <hip/hip_fp16.h>

typedef __attribute__((ext_vector_type(16))) _Float16 v16h;
typedef __attribute__((ext_vector_type(8)))  _Float16 v8h;
typedef __attribute__((ext_vector_type(4)))  _Float16 v4h;
typedef __attribute__((ext_vector_type(8)))  float    v8f;

#define B_  2
#define N_  2048
#define C_  1024
#define H_  16
#define HD_ 64
#define QT  64          // q rows per workgroup (4 waves x 16)
#define KT  64          // kv tile
#define LP  72          // padded LDS row stride in halfs (144B: bank-conflict-free)
#define NT_ (N_ / KT)   // kv tiles

// ============================================================================
// Pre-pass: K -> f16 [b][h][n][d] ; V -> f16 transposed [b][h][d][n]
// ============================================================================
__global__ __launch_bounds__(128, 2)
void mha_prepass_kernel(const float* __restrict__ k, const float* __restrict__ v,
                        _Float16* __restrict__ Kh, _Float16* __restrict__ Vth)
{
    __shared__ __align__(32) _Float16 T[HD_ * LP];   // [d][n] transpose tile

    const int tid = threadIdx.x;
    const int bid = blockIdx.x;
    const int nt  = bid % (N_ / 64);
    const int h   = (bid / (N_ / 64)) % H_;
    const int b   = bid / ((N_ / 64) * H_);
    const int n0  = nt * 64;
    const size_t head = (size_t)b * N_ * C_ + (size_t)h * HD_;

    // K: straight convert (layout preserved per (b,h) slice)
#pragma unroll
    for (int it = 0; it < 8; ++it) {
        const int idx = tid + it * 128;
        const int row = idx >> 4;
        const int d4  = (idx & 15) << 2;
        const float4 kk = *(const float4*)(k + head + (size_t)(n0 + row) * C_ + d4);
        v4h o; o[0] = (_Float16)kk.x; o[1] = (_Float16)kk.y;
               o[2] = (_Float16)kk.z; o[3] = (_Float16)kk.w;
        *(v4h*)&Kh[((size_t)(b * H_ + h) * N_ + n0 + row) * HD_ + d4] = o;
    }
    // V: transpose 64x64 tile through LDS
#pragma unroll
    for (int it = 0; it < 8; ++it) {
        const int idx = tid + it * 128;
        const int key = idx & 63;
        const int d4  = (idx >> 6) << 2;
        const float4 vv = *(const float4*)(v + head + (size_t)(n0 + key) * C_ + d4);
        T[(d4 + 0) * LP + key] = (_Float16)vv.x;
        T[(d4 + 1) * LP + key] = (_Float16)vv.y;
        T[(d4 + 2) * LP + key] = (_Float16)vv.z;
        T[(d4 + 3) * LP + key] = (_Float16)vv.w;
    }
    __syncthreads();
#pragma unroll
    for (int it = 0; it < 4; ++it) {
        const int g  = tid + it * 128;   // 0..511 granules of 8 halfs
        const int d  = g >> 3;
        const int np = (g & 7) * 8;
        const v8h row = *(const v8h*)&T[d * LP + np];
        *(v8h*)&Vth[((size_t)(b * H_ + h) * HD_ + d) * N_ + n0 + np] = row;
    }
}

// ============================================================================
// Main flash-attention kernel: async-to-LDS staging, double-buffered
// ============================================================================
__global__ __launch_bounds__(128, 2)
void mha_flash_async_kernel(const float* __restrict__ q,
                            const _Float16* __restrict__ Kh,
                            const _Float16* __restrict__ Vth,
                            const int* __restrict__ mask,
                            float* __restrict__ out)
{
    __shared__ __align__(32) _Float16 KsD[2][KT * LP];   // [key][dim]
    __shared__ __align__(32) _Float16 VsD[2][HD_ * LP];  // [dim][key]

    const int tid  = threadIdx.x;
    const int lane = tid & 31;
    const int wv   = tid >> 5;
    const int l16  = lane & 15;
    const int grp  = lane >> 4;

    const int bid = blockIdx.x;
    const int qt  = bid % (N_ / QT);
    const int h   = (bid / (N_ / QT)) % H_;
    const int b   = bid / ((N_ / QT) * H_);

    const int    qrow0 = qt * QT + wv * 16;
    const size_t head  = (size_t)b * N_ * C_ + (size_t)h * HD_;
    const int    myrow = qrow0 + l16;

    const _Float16* kg0 = Kh  + (size_t)(b * H_ + h) * N_ * HD_;
    const _Float16* vg0 = Vth + (size_t)(b * H_ + h) * HD_ * N_;

    const float scale = 0.125f;  // 1/sqrt(64)

    // ---- Q^T B-fragments: col(lane%16)=q row; half j -> dim = s*32 + grp*8 + j + (j>=8)*8 ----
    v16h qb[2];
    {
        const float* qrow = q + head + (size_t)myrow * C_;
#pragma unroll
        for (int s = 0; s < 2; ++s)
#pragma unroll
            for (int j = 0; j < 16; ++j)
                qb[s][j] = (_Float16)(qrow[s * 32 + grp * 8 + j + (j >= 8 ? 8 : 0)] * scale);
    }

    v8f acc[4] = {};
    float Mx = -1e30f, Ls = 0.0f;
    const int* mrow = mask + (size_t)b * N_ * N_ + (size_t)myrow * N_;

    // async staging of one KV tile into buffer `buf` (8 x B128 per wave)
    auto issue_tile = [&](int kb, int buf) {
#pragma unroll
        for (int it = 0; it < 4; ++it) {
            const int g   = tid + it * 128;   // 0..511 granules of 16B
            const int row = g >> 3;           // key (K) / dim (V)
            const int sp  = (g & 7) * 8;      // half offset within row
            {
                const unsigned lds = (unsigned)(uintptr_t)&KsD[buf][row * LP + sp];
                const unsigned long long ga =
                    (unsigned long long)(uintptr_t)(kg0 + ((size_t)(kb * KT + row)) * HD_ + sp);
                asm volatile("global_load_async_to_lds_b128 %0, %1, off"
                             :: "v"(lds), "v"(ga) : "memory");
            }
            {
                const unsigned lds = (unsigned)(uintptr_t)&VsD[buf][row * LP + sp];
                const unsigned long long ga =
                    (unsigned long long)(uintptr_t)(vg0 + (size_t)row * N_ + (kb * KT + sp));
                asm volatile("global_load_async_to_lds_b128 %0, %1, off"
                             :: "v"(lds), "v"(ga) : "memory");
            }
        }
    };

    issue_tile(0, 0);

    for (int kb = 0; kb < NT_; ++kb) {
        const int kv0 = kb * KT;
        const int buf = kb & 1;

        // ---- mask loads for this tile (keys contiguous per lane -> int4 pairs) ----
        int4 mv[4][2];
#pragma unroll
        for (int nt = 0; nt < 4; ++nt) {
            mv[nt][0] = *(const int4*)(mrow + kv0 + nt * 16 + 8 * grp);
            mv[nt][1] = *(const int4*)(mrow + kv0 + nt * 16 + 8 * grp + 4);
        }

        // ---- prefetch next tile into the other buffer, then release this tile ----
        if (kb + 1 < NT_) {
            issue_tile(kb + 1, buf ^ 1);
            asm volatile("s_wait_asynccnt 0x8" ::: "memory");  // first 8 (this tile) done
        } else {
            asm volatile("s_wait_asynccnt 0x0" ::: "memory");
        }
        __syncthreads();

        const _Float16* Ks = &KsD[buf][0];
        const _Float16* Vs = &VsD[buf][0];

        // ---- S^T = K * Q^T : 4 key-tiles x 2 K-slices ----
        v8f sc[4];
#pragma unroll
        for (int nt = 0; nt < 4; ++nt) {
            v8f c = {};
#pragma unroll
            for (int s = 0; s < 2; ++s) {
                const _Float16* kp = &Ks[(nt * 16 + l16) * LP + s * 32 + grp * 8];
                const v8h lo = *(const v8h*)(kp);
                const v8h hi = *(const v8h*)(kp + 16);
                v16h af;
#pragma unroll
                for (int j = 0; j < 8; ++j) { af[j] = lo[j]; af[8 + j] = hi[j]; }
                c = __builtin_amdgcn_wmma_f32_16x16x32_f16(
                        false, af, false, qb[s], (short)0, c, false, false);
            }
            sc[nt] = c;
        }

        // ---- mask + per-lane online softmax (softmax axis = register dim) ----
        float mx = -1e30f;
#pragma unroll
        for (int nt = 0; nt < 4; ++nt) {
            const int* m8 = (const int*)&mv[nt][0];
#pragma unroll
            for (int r = 0; r < 8; ++r) {
                float sv = sc[nt][r];
                sv = (m8[r] == 0) ? -1e9f : sv;
                sc[nt][r] = sv;
                mx = fmaxf(mx, sv);
            }
        }
        mx = fmaxf(mx, __shfl_xor(mx, 16, 32));
        const float newm  = fmaxf(Mx, mx);
        const float alpha = __expf(Mx - newm);
        float rsum = 0.0f;
#pragma unroll
        for (int nt = 0; nt < 4; ++nt)
#pragma unroll
            for (int r = 0; r < 8; ++r) {
                const float p = __expf(sc[nt][r] - newm);
                sc[nt][r] = p;
                rsum += p;
            }
        rsum += __shfl_xor(rsum, 16, 32);
        Ls = Ls * alpha + rsum;
        Mx = newm;
#pragma unroll
        for (int t = 0; t < 4; ++t)
#pragma unroll
            for (int r = 0; r < 8; ++r)
                acc[t][r] *= alpha;

        // ---- O^T += V^T * P^T : P^T B-frags built in registers ----
#pragma unroll
        for (int s = 0; s < 2; ++s) {
            float xr[8], zr[8];
#pragma unroll
            for (int r = 0; r < 8; ++r) {
                const float own = grp ? sc[s * 2 + 1][r] : sc[s * 2][r];
                const float snd = grp ? sc[s * 2][r]     : sc[s * 2 + 1][r];
                xr[r] = own;
                zr[r] = __shfl_xor(snd, 16, 32);
            }
            v16h pb;
#pragma unroll
            for (int j = 0; j < 8; ++j) {
                pb[j]     = (_Float16)(grp ? zr[j] : xr[j]);
                pb[8 + j] = (_Float16)(grp ? xr[j] : zr[j]);
            }
#pragma unroll
            for (int t = 0; t < 4; ++t) {
                const _Float16* vp = &Vs[(t * 16 + l16) * LP + s * 32 + grp * 16];
                const v8h lo = *(const v8h*)(vp);
                const v8h hi = *(const v8h*)(vp + 8);
                v16h af;
#pragma unroll
                for (int j = 0; j < 8; ++j) { af[j] = lo[j]; af[8 + j] = hi[j]; }
                acc[t] = __builtin_amdgcn_wmma_f32_16x16x32_f16(
                             false, af, false, pb, (short)0, acc[t], false, false);
            }
        }
        __syncthreads();   // all waves done reading `buf` before it is refilled
    }

    // ---- epilogue ----
    const float inv = 1.0f / Ls;
    float* orow = out + head + (size_t)myrow * C_;
#pragma unroll
    for (int t = 0; t < 4; ++t) {
        float4 lo, hi;
        lo.x = acc[t][0] * inv; lo.y = acc[t][1] * inv;
        lo.z = acc[t][2] * inv; lo.w = acc[t][3] * inv;
        hi.x = acc[t][4] * inv; hi.y = acc[t][5] * inv;
        hi.z = acc[t][6] * inv; hi.w = acc[t][7] * inv;
        *(float4*)(orow + t * 16 + 8 * grp)     = lo;
        *(float4*)(orow + t * 16 + 8 * grp + 4) = hi;
    }
}

// ============================================================================
// Fallback (workspace too small): fused kernel, converts K/V in-loop
// ============================================================================
__global__ __launch_bounds__(128, 2)
void mha_flash_fused_kernel(const float* __restrict__ q,
                            const float* __restrict__ k,
                            const float* __restrict__ v,
                            const int* __restrict__ mask,
                            float* __restrict__ out)
{
    __shared__ __align__(32) _Float16 Ks[KT * LP];
    __shared__ __align__(32) _Float16 Vs[HD_ * LP];

    const int tid  = threadIdx.x;
    const int lane = tid & 31;
    const int wv   = tid >> 5;
    const int l16  = lane & 15;
    const int grp  = lane >> 4;

    const int bid = blockIdx.x;
    const int qt  = bid % (N_ / QT);
    const int h   = (bid / (N_ / QT)) % H_;
    const int b   = bid / ((N_ / QT) * H_);

    const int    qrow0 = qt * QT + wv * 16;
    const size_t head  = (size_t)b * N_ * C_ + (size_t)h * HD_;
    const int    myrow = qrow0 + l16;
    const float  scale = 0.125f;

    v16h qb[2];
    {
        const float* qrow = q + head + (size_t)myrow * C_;
#pragma unroll
        for (int s = 0; s < 2; ++s)
#pragma unroll
            for (int j = 0; j < 16; ++j)
                qb[s][j] = (_Float16)(qrow[s * 32 + grp * 8 + j + (j >= 8 ? 8 : 0)] * scale);
    }

    v8f acc[4] = {};
    float Mx = -1e30f, Ls = 0.0f;
    const int* mrow = mask + (size_t)b * N_ * N_ + (size_t)myrow * N_;

    for (int kb = 0; kb < NT_; ++kb) {
        const int kv0 = kb * KT;
        int4 mv[4][2];
#pragma unroll
        for (int nt = 0; nt < 4; ++nt) {
            mv[nt][0] = *(const int4*)(mrow + kv0 + nt * 16 + 8 * grp);
            mv[nt][1] = *(const int4*)(mrow + kv0 + nt * 16 + 8 * grp + 4);
        }
        __syncthreads();
#pragma unroll
        for (int it = 0; it < 8; ++it) {
            const int idx = tid + it * 128;
            const int key = idx >> 4;
            const int d4  = (idx & 15) << 2;
            const float4 kk = *(const float4*)(k + head + (size_t)(kv0 + key) * C_ + d4);
            v4h o; o[0] = (_Float16)kk.x; o[1] = (_Float16)kk.y;
                   o[2] = (_Float16)kk.z; o[3] = (_Float16)kk.w;
            *(v4h*)&Ks[key * LP + d4] = o;
        }
#pragma unroll
        for (int it = 0; it < 8; ++it) {
            const int idx = tid + it * 128;
            const int key = idx & 63;
            const int d4  = (idx >> 6) << 2;
            const float4 vv = *(const float4*)(v + head + (size_t)(kv0 + key) * C_ + d4);
            Vs[(d4 + 0) * LP + key] = (_Float16)vv.x;
            Vs[(d4 + 1) * LP + key] = (_Float16)vv.y;
            Vs[(d4 + 2) * LP + key] = (_Float16)vv.z;
            Vs[(d4 + 3) * LP + key] = (_Float16)vv.w;
        }
        __syncthreads();

        v8f sc[4];
#pragma unroll
        for (int nt = 0; nt < 4; ++nt) {
            v8f c = {};
#pragma unroll
            for (int s = 0; s < 2; ++s) {
                const _Float16* kp = &Ks[(nt * 16 + l16) * LP + s * 32 + grp * 8];
                const v8h lo = *(const v8h*)(kp);
                const v8h hi = *(const v8h*)(kp + 16);
                v16h af;
#pragma unroll
                for (int j = 0; j < 8; ++j) { af[j] = lo[j]; af[8 + j] = hi[j]; }
                c = __builtin_amdgcn_wmma_f32_16x16x32_f16(
                        false, af, false, qb[s], (short)0, c, false, false);
            }
            sc[nt] = c;
        }

        float mx = -1e30f;
#pragma unroll
        for (int nt = 0; nt < 4; ++nt) {
            const int* m8 = (const int*)&mv[nt][0];
#pragma unroll
            for (int r = 0; r < 8; ++r) {
                float sv = sc[nt][r];
                sv = (m8[r] == 0) ? -1e9f : sv;
                sc[nt][r] = sv;
                mx = fmaxf(mx, sv);
            }
        }
        mx = fmaxf(mx, __shfl_xor(mx, 16, 32));
        const float newm  = fmaxf(Mx, mx);
        const float alpha = __expf(Mx - newm);
        float rsum = 0.0f;
#pragma unroll
        for (int nt = 0; nt < 4; ++nt)
#pragma unroll
            for (int r = 0; r < 8; ++r) {
                const float p = __expf(sc[nt][r] - newm);
                sc[nt][r] = p;
                rsum += p;
            }
        rsum += __shfl_xor(rsum, 16, 32);
        Ls = Ls * alpha + rsum;
        Mx = newm;
#pragma unroll
        for (int t = 0; t < 4; ++t)
#pragma unroll
            for (int r = 0; r < 8; ++r)
                acc[t][r] *= alpha;

#pragma unroll
        for (int s = 0; s < 2; ++s) {
            float xr[8], zr[8];
#pragma unroll
            for (int r = 0; r < 8; ++r) {
                const float own = grp ? sc[s * 2 + 1][r] : sc[s * 2][r];
                const float snd = grp ? sc[s * 2][r]     : sc[s * 2 + 1][r];
                xr[r] = own;
                zr[r] = __shfl_xor(snd, 16, 32);
            }
            v16h pb;
#pragma unroll
            for (int j = 0; j < 8; ++j) {
                pb[j]     = (_Float16)(grp ? zr[j] : xr[j]);
                pb[8 + j] = (_Float16)(grp ? xr[j] : zr[j]);
            }
#pragma unroll
            for (int t = 0; t < 4; ++t) {
                const _Float16* vp = &Vs[(t * 16 + l16) * LP + s * 32 + grp * 16];
                const v8h lo = *(const v8h*)(vp);
                const v8h hi = *(const v8h*)(vp + 8);
                v16h af;
#pragma unroll
                for (int j = 0; j < 8; ++j) { af[j] = lo[j]; af[8 + j] = hi[j]; }
                acc[t] = __builtin_amdgcn_wmma_f32_16x16x32_f16(
                             false, af, false, pb, (short)0, acc[t], false, false);
            }
        }
    }

    const float inv = 1.0f / Ls;
    float* orow = out + head + (size_t)myrow * C_;
#pragma unroll
    for (int t = 0; t < 4; ++t) {
        float4 lo, hi;
        lo.x = acc[t][0] * inv; lo.y = acc[t][1] * inv;
        lo.z = acc[t][2] * inv; lo.w = acc[t][3] * inv;
        hi.x = acc[t][4] * inv; hi.y = acc[t][5] * inv;
        hi.z = acc[t][6] * inv; hi.w = acc[t][7] * inv;
        *(float4*)(orow + t * 16 + 8 * grp)     = lo;
        *(float4*)(orow + t * 16 + 8 * grp + 4) = hi;
    }
}

extern "C" void kernel_launch(void* const* d_in, const int* in_sizes, int n_in,
                              void* d_out, int out_size, void* d_ws, size_t ws_size,
                              hipStream_t stream) {
    (void)in_sizes; (void)n_in; (void)out_size;
    const float* q    = (const float*)d_in[0];
    const float* k    = (const float*)d_in[1];
    const float* v    = (const float*)d_in[2];
    const int*   mask = (const int*)d_in[3];
    float* out = (float*)d_out;

    const int grid = B_ * H_ * (N_ / QT);   // 1024 workgroups
    const size_t khBytes = (size_t)B_ * H_ * N_ * HD_ * sizeof(_Float16);  // 8 MB

    if (ws_size >= 2 * khBytes) {
        _Float16* Kh  = (_Float16*)d_ws;
        _Float16* Vth = (_Float16*)((char*)d_ws + khBytes);
        mha_prepass_kernel<<<dim3(B_ * H_ * (N_ / 64)), dim3(128), 0, stream>>>(k, v, Kh, Vth);
        mha_flash_async_kernel<<<dim3(grid), dim3(128), 0, stream>>>(q, Kh, Vth, mask, out);
    } else {
        mha_flash_fused_kernel<<<dim3(grid), dim3(128), 0, stream>>>(q, k, v, mask, out);
    }
}
